// DynamicMoEModel_42116449304981
// MI455X (gfx1250) — compile-verified, matrix operand order checked by reading, not run on hardware
//
#include <hip/hip_runtime.h>
#include <hip/hip_bf16.h>
#include <stdint.h>

// ---------------------------------------------------------------------------
// CDNA5 (gfx1250) wave32 WMMA implementation of the DynamicMoE forward pass.
// All large contractions use V_WMMA_F32_16X16X32_BF16 (bf16 in, f32 acc).
// ---------------------------------------------------------------------------

#define DEVI __device__ __forceinline__

typedef __bf16 v16bf __attribute__((ext_vector_type(16)));
typedef float  v8f   __attribute__((ext_vector_type(8)));
typedef uint32_t u32x4 __attribute__((ext_vector_type(4)));

static constexpr int  Bb = 2, Tt = 1024, Cc = 256, Ff = 1024, Ee = 32, Vv = 50257, Hh = 8, Dd = 32;
static constexpr int  Nn = Bb * Tt;           // 2048 tokens
static constexpr int  VP = 50304;             // vocab padded to 64*786 (4 N-tiles/wave)
static constexpr float NEGF = -3.3895314e38f; // -bf16 max

union FragU { u32x4 u[2]; v16bf v; };

DEVI uint16_t f2bf(float f) {
    uint32_t u = __float_as_uint(f);
    uint32_t r = (u + 0x7FFFu + ((u >> 16) & 1u)) >> 16;
    return (uint16_t)r;
}
DEVI float bf2f(uint16_t h) { return __uint_as_float(((uint32_t)h) << 16); }

// 16x32 bf16 fragment (A operand, or B via the [N][K] transposed weights)
// from row-major storage with leading dimension ld. CDNA5 16-bit layout:
// lane&15 = row, lane>>4 selects the interleaved 8-wide K chunks.
DEVI v16bf load_frag(const uint16_t* base, int ld) {
    int lane = threadIdx.x & 31;
    int r = lane & 15, hi = lane >> 4;
    const uint16_t* p = base + (long)r * ld + hi * 8;
    FragU f;
    f.u[0] = *reinterpret_cast<const u32x4*>(p);
    f.u[1] = *reinterpret_cast<const u32x4*>(p + 16);
    return f.v;
}

DEVI v8f wmma_bf16(v16bf a, v16bf b, v8f c) {
    return __builtin_amdgcn_wmma_f32_16x16x32_bf16(false, a, false, b, (short)0, c, false, false);
}

// ---------------------------------------------------------------------------
// Weight conversion: f32 [rows][cols] -> bf16 transposed [cols][rows]
// ---------------------------------------------------------------------------
__global__ void k_transpose_bf16(const float* __restrict__ src, uint16_t* __restrict__ dst,
                                 int rows, int cols, long sstride, long dstride) {
    long bz = blockIdx.z;
    const float* s = src + bz * sstride;
    uint16_t* d = dst + bz * dstride;
    int idx = blockIdx.x * 256 + threadIdx.x;
    if (idx < rows * cols) {
        int r = idx / cols, c = idx % cols;
        d[(long)c * rows + r] = f2bf(s[(long)r * cols + c]);
    }
}

// lm_head [C][V] -> bf16 [VP][C], zero padded rows
__global__ void k_lmhead_t(const float* __restrict__ src, uint16_t* __restrict__ dst) {
    int idx = blockIdx.x * 256 + threadIdx.x;
    if (idx >= VP * Cc) return;
    int v = idx / Cc, c = idx % Cc;
    float x = (v < Vv) ? src[(long)c * Vv + v] : 0.f;
    dst[idx] = f2bf(x);
}

// ---------------------------------------------------------------------------
// Embedding + LayerNorm1  (block = 1 token, 256 threads)
// ---------------------------------------------------------------------------
__global__ void k_embed_ln(const int* __restrict__ ids, const float* __restrict__ emb,
                           const float* __restrict__ w, const float* __restrict__ b,
                           float* __restrict__ hs, uint16_t* __restrict__ xout) {
    int t = blockIdx.x, c = threadIdx.x;
    __shared__ float red[256];
    float x = emb[(long)ids[t] * Cc + c];
    hs[t * Cc + c] = x;
    red[c] = x; __syncthreads();
    for (int s = 128; s > 0; s >>= 1) { if (c < s) red[c] += red[c + s]; __syncthreads(); }
    float mu = red[0] * (1.f / Cc);
    __syncthreads();
    float dv = x - mu; red[c] = dv * dv; __syncthreads();
    for (int s = 128; s > 0; s >>= 1) { if (c < s) red[c] += red[c + s]; __syncthreads(); }
    float var = red[0] * (1.f / Cc);
    xout[t * Cc + c] = f2bf(dv * rsqrtf(var + 1e-5f) * w[c] + b[c]);
}

// LayerNorm over f32 input -> bf16
__global__ void k_ln(const float* __restrict__ src, const float* __restrict__ w,
                     const float* __restrict__ b, uint16_t* __restrict__ out) {
    int t = blockIdx.x, c = threadIdx.x;
    __shared__ float red[256];
    float x = src[t * Cc + c];
    red[c] = x; __syncthreads();
    for (int s = 128; s > 0; s >>= 1) { if (c < s) red[c] += red[c + s]; __syncthreads(); }
    float mu = red[0] * (1.f / Cc);
    __syncthreads();
    float dv = x - mu; red[c] = dv * dv; __syncthreads();
    for (int s = 128; s > 0; s >>= 1) { if (c < s) red[c] += red[c + s]; __syncthreads(); }
    float var = red[0] * (1.f / Cc);
    out[t * Cc + c] = f2bf(dv * rsqrtf(var + 1e-5f) * w[c] + b[c]);
}

// ---------------------------------------------------------------------------
// Fused QKV projection: one wave per 16x16 tile; A fragment shared by the
// three WMMAs per K step. Q/K scattered to [B,H,T,D], V to [B,H,D,T].
// ---------------------------------------------------------------------------
__global__ void k_qkv(const uint16_t* __restrict__ A,
                      const uint16_t* __restrict__ wqt, const uint16_t* __restrict__ wkt,
                      const uint16_t* __restrict__ wvt,
                      uint16_t* __restrict__ qb, uint16_t* __restrict__ kb,
                      uint16_t* __restrict__ vtb) {
    int m0 = blockIdx.x * 16, n0 = blockIdx.y * 16;
    int lane = threadIdx.x & 31;
    int colL = lane & 15, rowB = (lane >> 4) * 8;
    v8f aq = {0, 0, 0, 0, 0, 0, 0, 0}, ak = aq, av = aq;
    for (int k0 = 0; k0 < Cc; k0 += 32) {
        v16bf a = load_frag(A + m0 * Cc + k0, Cc);
        v16bf b0 = load_frag(wqt + n0 * Cc + k0, Cc);
        v16bf b1 = load_frag(wkt + n0 * Cc + k0, Cc);
        v16bf b2 = load_frag(wvt + n0 * Cc + k0, Cc);
        aq = wmma_bf16(a, b0, aq);
        ak = wmma_bf16(a, b1, ak);
        av = wmma_bf16(a, b2, av);
    }
    int col = n0 + colL;
    int h = col >> 5, d = col & 31;
#pragma unroll
    for (int r = 0; r < 8; r++) {
        int n = m0 + rowB + r;
        int b_ = n >> 10, tt_ = n & 1023;
        long qki = (((long)(b_ * Hh + h) * Tt + tt_) << 5) + d;
        qb[qki] = f2bf(aq[r]);
        kb[qki] = f2bf(ak[r]);
        vtb[(((long)(b_ * Hh + h) * Dd + d) << 10) + tt_] = f2bf(av[r]);
    }
}

// Output projection + residual: hs = hs + ao @ wo (C fragment = hs)
__global__ void k_oproj(const uint16_t* __restrict__ A, const uint16_t* __restrict__ Bt,
                        float* __restrict__ hs) {
    int m0 = blockIdx.x * 16, n0 = blockIdx.y * 16;
    int lane = threadIdx.x & 31;
    int colL = lane & 15, rowB = (lane >> 4) * 8;
    v8f acc;
#pragma unroll
    for (int r = 0; r < 8; r++) acc[r] = hs[(m0 + rowB + r) * Cc + n0 + colL];
    for (int k0 = 0; k0 < Cc; k0 += 32) {
        v16bf a = load_frag(A + m0 * Cc + k0, Cc);
        v16bf b = load_frag(Bt + n0 * Cc + k0, Cc);
        acc = wmma_bf16(a, b, acc);
    }
#pragma unroll
    for (int r = 0; r < 8; r++) hs[(m0 + rowB + r) * Cc + n0 + colL] = acc[r];
}

// ---------------------------------------------------------------------------
// Flash-style causal attention, one wave per (b*H+h, 16-query tile).
// S = Q K^T via WMMA. Online softmax is computed WITHOUT cross-lane shuffles:
// the C-frag lanes spill raw S to LDS; lanes 0..15 (lane == row) own the
// per-row running (m,l) state in registers, apply mask/exp in-lane, and emit
// packed-bf16 P rows plus per-row rescale factors through LDS. P is then
// consumed as an A fragment for the P·V WMMAs (V stored transposed).
// ---------------------------------------------------------------------------
__global__ void k_attn(const uint16_t* __restrict__ qb, const uint16_t* __restrict__ kb,
                       const uint16_t* __restrict__ vtb, uint16_t* __restrict__ ao) {
    int bh = blockIdx.x;          // 0..15
    int q0 = blockIdx.y * 16;
    int lane = threadIdx.x & 31;
    int colL = lane & 15, rowB = (lane >> 4) * 8;
    __shared__ __align__(16) float    sbuf[16][32];
    __shared__ __align__(16) uint16_t pbuf[16][40];
    __shared__ __align__(16) float    scbuf[16];
    __shared__ __align__(16) float    lbuf[16];
    const float rs = 0.17677669529663689f;  // 1/sqrt(32)

    v16bf aq = load_frag(qb + (bh * Tt + q0) * Dd, Dd);  // full K = D = 32
    v8f acc0 = {0, 0, 0, 0, 0, 0, 0, 0}, acc1 = {0, 0, 0, 0, 0, 0, 0, 0};
    float mrow = -3.0e38f, lrow = 0.f;  // per-row state, valid in lanes 0..15

    for (int j0 = 0; j0 < q0 + 16; j0 += 32) {
        v16bf bk0 = load_frag(kb + (bh * Tt + j0) * Dd, Dd);
        v16bf bk1 = load_frag(kb + (bh * Tt + j0 + 16) * Dd, Dd);
        v8f z = {0, 0, 0, 0, 0, 0, 0, 0};
        v8f s0 = wmma_bf16(aq, bk0, z);
        v8f s1 = wmma_bf16(aq, bk1, z);
#pragma unroll
        for (int r = 0; r < 8; r++) {
            sbuf[rowB + r][colL] = s0[r];
            sbuf[rowB + r][16 + colL] = s1[r];
        }
        __syncthreads();
        if (lane < 16) {  // per-row online softmax, fully in-lane
            int row = q0 + lane;
            float xv[32];
            float mx = -3.0e38f;
#pragma unroll
            for (int j = 0; j < 32; j++) {
                float x = sbuf[lane][j] * rs;
                x = (j0 + j <= row) ? x : NEGF;
                xv[j] = x;
                mx = fmaxf(mx, x);
            }
            float mnew = fmaxf(mrow, mx);
            float sc = __expf(mrow - mnew);
            float sum = 0.f;
#pragma unroll
            for (int j = 0; j < 32; j += 2) {
                float p0 = __expf(xv[j] - mnew);
                float p1 = __expf(xv[j + 1] - mnew);
                sum += p0 + p1;
                *(uint32_t*)&pbuf[lane][j] = (uint32_t)f2bf(p0) | ((uint32_t)f2bf(p1) << 16);
            }
            lrow = lrow * sc + sum;
            mrow = mnew;
            scbuf[lane] = sc;
        }
        __syncthreads();
#pragma unroll
        for (int r = 0; r < 8; r++) {  // rescale accumulators by row factor
            float s = scbuf[rowB + r];
            acc0[r] *= s;
            acc1[r] *= s;
        }
        v16bf pa = load_frag(&pbuf[0][0], 40);
        v16bf bv0 = load_frag(vtb + (bh * Dd + 0) * Tt + j0, Tt);
        v16bf bv1 = load_frag(vtb + (bh * Dd + 16) * Tt + j0, Tt);
        acc0 = wmma_bf16(pa, bv0, acc0);
        acc1 = wmma_bf16(pa, bv1, acc1);
        __syncthreads();
    }
    if (lane < 16) lbuf[lane] = lrow;
    __syncthreads();
    int b_ = bh >> 3, h = bh & 7;
#pragma unroll
    for (int r = 0; r < 8; r++) {
        int n = b_ * Tt + q0 + rowB + r;
        float inv = 1.f / lbuf[rowB + r];
        ao[n * Cc + h * Dd + colL] = f2bf(acc0[r] * inv);
        ao[n * Cc + h * Dd + 16 + colL] = f2bf(acc1[r] * inv);
    }
}

// ---------------------------------------------------------------------------
// sim_matrix column inverse norms
// ---------------------------------------------------------------------------
__global__ void k_simnorm(const float* __restrict__ sim, float* __restrict__ snorm) {
    int e = threadIdx.x;  // 32 lanes
    float s = 0.f;
    for (int c = 0; c < Cc; c++) { float v = sim[c * Ee + e]; s += v * v; }
    snorm[e] = 1.f / fmaxf(sqrtf(s), 1e-12f);
}

// ---------------------------------------------------------------------------
// Gating: one wave per token (lane = expert). Writes pre + activation_mask to
// d_out and routing_weights to scratch.
// ---------------------------------------------------------------------------
__global__ void k_gate(const uint16_t* __restrict__ xt, const float* __restrict__ sim,
                       const float* __restrict__ snorm, const float* __restrict__ gates,
                       float* __restrict__ pre_out, float* __restrict__ amask_out,
                       float* __restrict__ rw) {
    int t = blockIdx.x, e = threadIdx.x;
    __shared__ float xr[256];
    __shared__ float lsh[32];
    for (int i = 0; i < 8; i++) { int c = e + i * 32; xr[c] = bf2f(xt[t * Cc + c]); }
    __syncthreads();
    float sq = 0.f;
    for (int i = 0; i < 8; i++) { float v = xr[e + i * 32]; sq += v * v; }
    for (int m = 1; m < 32; m <<= 1) sq += __shfl_xor(sq, m, 32);
    float invx = 1.f / fmaxf(sqrtf(sq), 1e-12f);
    float dot = 0.f;
    for (int c = 0; c < Cc; c++) dot += xr[c] * sim[c * Ee + e];
    float logit = dot * invx * snorm[e];
    float prev = logit - 1.f / (1.f + __expf(-gates[e]));
    float gated = fmaxf(prev, 0.f);
    float mask = (gated > 0.f) ? 1.f : 0.f;
    pre_out[t * Ee + e] = prev;
    int active = (int)__popcll(__ballot(mask > 0.f));
    float am = mask;
    lsh[e] = logit;
    __syncthreads();
    if (active == 0) {  // top-(E/2) fallback by rank
        int rank = 0;
        for (int j = 0; j < Ee; j++) {
            float lj = lsh[j];
            if (lj > logit || (lj == logit && j < e)) rank++;
        }
        am = (rank < Ee / 2) ? 1.f : 0.f;
    }
    amask_out[t * Ee + e] = am;
    float gl = (am > 0.f) ? gated : NEGF;
    float mx = gl;
    for (int m = 1; m < 32; m <<= 1) mx = fmaxf(mx, __shfl_xor(mx, m, 32));
    float ex = __expf(gl - mx);
    float sm = ex;
    for (int m = 1; m < 32; m <<= 1) sm += __shfl_xor(sm, m, 32);
    rw[t * Ee + e] = ex / sm;
}

__global__ void k_zero(float* __restrict__ p, int n) {
    int i = blockIdx.x * 256 + threadIdx.x;
    if (i < n) p[i] = 0.f;
}

// ---------------------------------------------------------------------------
// Fused MoE expert FFN. Block = (16-token tile, expert), 8 waves.
// Stage 1: h = gelu(xt @ w1[e]) -> 16x1024 bf16 LDS tile; the 8 A fragments
//          of the token tile are hoisted into registers and reused for all
//          8 N-tiles handled by the wave.
// Stage 2: eo = h @ w2[e]; each LDS A fragment feeds 2 WMMAs. Masked write of
//          full_expert_outputs + routed accumulation via f32 global atomics.
// ---------------------------------------------------------------------------
__global__ void k_moe(const uint16_t* __restrict__ xt, const uint16_t* __restrict__ w1t,
                      const uint16_t* __restrict__ w2t, const float* __restrict__ rw,
                      const float* __restrict__ amask, float* __restrict__ feo,
                      float* __restrict__ moeacc) {
    int m0 = blockIdx.x * 16;
    int e = blockIdx.y;
    int wave = threadIdx.x >> 5;
    int lane = threadIdx.x & 31;
    int colL = lane & 15, rowB = (lane >> 4) * 8;
    __shared__ __align__(16) uint16_t hbuf[16][1032];

    const uint16_t* w1e = w1t + (long)e * Ff * Cc;  // [f][c]
    v16bf af[8];
#pragma unroll
    for (int kk = 0; kk < 8; kk++) af[kk] = load_frag(xt + m0 * Cc + kk * 32, Cc);
#pragma unroll
    for (int nt = 0; nt < 8; nt++) {
        int n0 = wave * 128 + nt * 16;
        v8f acc = {0, 0, 0, 0, 0, 0, 0, 0};
#pragma unroll
        for (int kk = 0; kk < 8; kk++) {
            v16bf b = load_frag(w1e + n0 * Cc + kk * 32, Cc);
            acc = wmma_bf16(af[kk], b, acc);
        }
#pragma unroll
        for (int r = 0; r < 8; r++) {
            float x = acc[r];
            float g = 0.5f * x * (1.f + erff(x * 0.70710678118654752f));  // exact gelu
            hbuf[rowB + r][n0 + colL] = f2bf(g);
        }
    }
    __syncthreads();

    const uint16_t* w2e = w2t + (long)e * Cc * Ff;  // [c][f]
    float amv[8], rwv[8];
#pragma unroll
    for (int r = 0; r < 8; r++) {
        int tok = m0 + rowB + r;
        amv[r] = amask[tok * Ee + e];
        rwv[r] = rw[tok * Ee + e];
    }
    int n0a = wave * 32;
    v8f acc0 = {0, 0, 0, 0, 0, 0, 0, 0}, acc1 = {0, 0, 0, 0, 0, 0, 0, 0};
    for (int k0 = 0; k0 < Ff; k0 += 32) {
        v16bf a = load_frag(&hbuf[0][0] + k0, 1032);  // from LDS
        v16bf b0 = load_frag(w2e + n0a * Ff + k0, Ff);
        v16bf b1 = load_frag(w2e + (n0a + 16) * Ff + k0, Ff);
        acc0 = wmma_bf16(a, b0, acc0);
        acc1 = wmma_bf16(a, b1, acc1);
    }
#pragma unroll
    for (int i = 0; i < 2; i++) {
        v8f acc = (i == 0) ? acc0 : acc1;
        int col = n0a + i * 16 + colL;
#pragma unroll
        for (int r = 0; r < 8; r++) {
            int tok = m0 + rowB + r;
            float val = amv[r] * acc[r];
            feo[((long)tok * Ee + e) * Cc + col] = val;
            atomicAdd(&moeacc[tok * Cc + col], rwv[r] * val);
        }
    }
}

__global__ void k_addcvt(const float* __restrict__ hs, const float* __restrict__ moe,
                         uint16_t* __restrict__ hsf) {
    int i = blockIdx.x * 256 + threadIdx.x;
    if (i < Nn * Cc) hsf[i] = f2bf(hs[i] + moe[i]);
}

// ---------------------------------------------------------------------------
// lm_head GEMM: [2048x256] x [256xVP] -> f32 logits. One wave per 16x64
// output strip: the A fragment is reused across 4 back-to-back WMMAs.
// ---------------------------------------------------------------------------
__global__ void k_lmhead(const uint16_t* __restrict__ A, const uint16_t* __restrict__ Bt,
                         float* __restrict__ out) {
    int m0 = blockIdx.x * 16, n0 = blockIdx.y * 64;
    int lane = threadIdx.x & 31;
    int colL = lane & 15, rowB = (lane >> 4) * 8;
    v8f acc[4];
#pragma unroll
    for (int j = 0; j < 4; j++) acc[j] = v8f{0, 0, 0, 0, 0, 0, 0, 0};
    for (int k0 = 0; k0 < Cc; k0 += 32) {
        __builtin_prefetch(Bt + (long)n0 * Cc + k0 + 4096, 0, 1);  // global_prefetch_b8
        v16bf a = load_frag(A + m0 * Cc + k0, Cc);
#pragma unroll
        for (int j = 0; j < 4; j++) {
            v16bf b = load_frag(Bt + (long)(n0 + j * 16) * Cc + k0, Cc);
            acc[j] = wmma_bf16(a, b, acc[j]);
        }
    }
#pragma unroll
    for (int j = 0; j < 4; j++) {
        int col = n0 + j * 16 + colL;
        if (col < Vv) {
#pragma unroll
            for (int r = 0; r < 8; r++) out[(long)(m0 + rowB + r) * Vv + col] = acc[j][r];
        }
    }
}

// ---------------------------------------------------------------------------
extern "C" void kernel_launch(void* const* d_in, const int* in_sizes, int n_in,
                              void* d_out, int out_size, void* d_ws, size_t ws_size,
                              hipStream_t stream) {
    (void)in_sizes; (void)n_in; (void)out_size; (void)ws_size;
    const int*   ids     = (const int*)d_in[0];
    const float* emb     = (const float*)d_in[1];
    const float* wq      = (const float*)d_in[2];
    const float* wk      = (const float*)d_in[3];
    const float* wv      = (const float*)d_in[4];
    const float* wo      = (const float*)d_in[5];
    const float* ln1w    = (const float*)d_in[6];
    const float* ln1b    = (const float*)d_in[7];
    const float* ln2w    = (const float*)d_in[8];
    const float* ln2b    = (const float*)d_in[9];
    const float* sim     = (const float*)d_in[10];
    const float* gates   = (const float*)d_in[11];
    const float* w1      = (const float*)d_in[12];
    const float* w2      = (const float*)d_in[13];
    const float* lm_head = (const float*)d_in[14];

    // workspace carve-out (deterministic layout every call)
    char* ws = (char*)d_ws;
    auto carve = [&](size_t bytes) -> char* {
        char* p = ws;
        ws += (bytes + 255) & ~(size_t)255;
        return p;
    };
    uint16_t* wqt  = (uint16_t*)carve((size_t)Cc * Cc * 2);
    uint16_t* wkt  = (uint16_t*)carve((size_t)Cc * Cc * 2);
    uint16_t* wvt  = (uint16_t*)carve((size_t)Cc * Cc * 2);
    uint16_t* wot  = (uint16_t*)carve((size_t)Cc * Cc * 2);
    uint16_t* w1t  = (uint16_t*)carve((size_t)Ee * Ff * Cc * 2);
    uint16_t* w2t  = (uint16_t*)carve((size_t)Ee * Cc * Ff * 2);
    uint16_t* lmht = (uint16_t*)carve((size_t)VP * Cc * 2);
    float*    hs   = (float*)carve((size_t)Nn * Cc * 4);
    uint16_t* xln  = (uint16_t*)carve((size_t)Nn * Cc * 2);
    uint16_t* qb   = (uint16_t*)carve((size_t)Nn * Cc * 2);
    uint16_t* kb   = (uint16_t*)carve((size_t)Nn * Cc * 2);
    uint16_t* vtb  = (uint16_t*)carve((size_t)Nn * Cc * 2);
    uint16_t* aob  = (uint16_t*)carve((size_t)Nn * Cc * 2);
    uint16_t* xtb  = (uint16_t*)carve((size_t)Nn * Cc * 2);
    float*    moeb = (float*)carve((size_t)Nn * Cc * 4);
    float*    rwb  = (float*)carve((size_t)Nn * Ee * 4);
    float*    snb  = (float*)carve(128);
    uint16_t* hsf  = (uint16_t*)carve((size_t)Nn * Cc * 2);

    // output regions: (out_logits, full_expert_outputs, pre, activation_mask)
    float* outF       = (float*)d_out;
    float* out_logits = outF;
    float* out_feo    = outF + (long)Nn * Vv;
    float* out_pre    = out_feo + (long)Nn * Ee * Cc;
    float* out_amask  = out_pre + (long)Nn * Ee;

    dim3 b256(256);
    // weight conversion / transpose to bf16 [N][K]
    int nCC = Cc * Cc;
    k_transpose_bf16<<<dim3((nCC + 255) / 256, 1, 1), b256, 0, stream>>>(wq, wqt, Cc, Cc, 0L, 0L);
    k_transpose_bf16<<<dim3((nCC + 255) / 256, 1, 1), b256, 0, stream>>>(wk, wkt, Cc, Cc, 0L, 0L);
    k_transpose_bf16<<<dim3((nCC + 255) / 256, 1, 1), b256, 0, stream>>>(wv, wvt, Cc, Cc, 0L, 0L);
    k_transpose_bf16<<<dim3((nCC + 255) / 256, 1, 1), b256, 0, stream>>>(wo, wot, Cc, Cc, 0L, 0L);
    int nCF = Cc * Ff;
    k_transpose_bf16<<<dim3((nCF + 255) / 256, 1, Ee), b256, 0, stream>>>(w1, w1t, Cc, Ff, (long)Cc * Ff, (long)Ff * Cc);
    k_transpose_bf16<<<dim3((nCF + 255) / 256, 1, Ee), b256, 0, stream>>>(w2, w2t, Ff, Cc, (long)Ff * Cc, (long)Cc * Ff);
    k_lmhead_t<<<dim3((VP * Cc + 255) / 256), b256, 0, stream>>>(lm_head, lmht);

    // embed + LN1
    k_embed_ln<<<dim3(Nn), b256, 0, stream>>>(ids, emb, ln1w, ln1b, hs, xln);

    // fused QKV projection (WMMA, shared A fragments)
    dim3 g16(Nn / 16, Cc / 16);
    k_qkv<<<g16, 32, 0, stream>>>(xln, wqt, wkt, wvt, qb, kb, vtb);

    // causal attention (WMMA flash, shuffle-free online softmax)
    k_attn<<<dim3(Bb * Hh, Tt / 16), 32, 0, stream>>>(qb, kb, vtb, aob);

    // output projection + residual (WMMA, C-frag = hs)
    k_oproj<<<g16, 32, 0, stream>>>(aob, wot, hs);

    // LN2 + gating
    k_ln<<<dim3(Nn), b256, 0, stream>>>(hs, ln2w, ln2b, xtb);
    k_simnorm<<<dim3(1), 32, 0, stream>>>(sim, snb);
    k_gate<<<dim3(Nn), 32, 0, stream>>>(xtb, sim, snb, gates, out_pre, out_amask, rwb);

    // MoE (WMMA, fused up/act/down with LDS staging)
    k_zero<<<dim3((Nn * Cc + 255) / 256), b256, 0, stream>>>(moeb, Nn * Cc);
    k_moe<<<dim3(Nn / 16, Ee), b256, 0, stream>>>(xtb, w1t, w2t, rwb, out_amask, out_feo, moeb);

    // residual + lm_head GEMM (WMMA, 4 N-tiles per wave)
    k_addcvt<<<dim3((Nn * Cc + 255) / 256), b256, 0, stream>>>(hs, moeb, hsf);
    k_lmhead<<<dim3(Nn / 16, VP / 64), 32, 0, stream>>>(hsf, lmht, out_logits);
}
